// StreamingMultiScaleTCN_33792802684961
// MI455X (gfx1250) — matrix operand order repaired and checked by reading, not compiled, not run on hardware
//
#include <hip/hip_runtime.h>

typedef unsigned short u16;
typedef unsigned int   u32;
typedef unsigned long long u64;

typedef __attribute__((ext_vector_type(16))) __bf16 v16bf;
typedef __attribute__((ext_vector_type(8)))  float  v8f;

#define T_LEN  8192
#define TILE_T 128
#define HALO   64
#define ROWS   (TILE_T + HALO)   // 192 rows of x (incl. causal halo)
#define XS     72                // x LDS row stride in ushorts (64 + pad, bank-conflict free)
#define XCS    136               // xc LDS row stride in ushorts (128 + pad)

union FragBF {
  v16bf v;
  u16   u[16];
  uint4 q[2];
};

__device__ __forceinline__ u16 f2bf(float f) {
  u32 u = __float_as_uint(f);
  u += 0x7fffu + ((u >> 16) & 1u);   // round-to-nearest-even bf16
  return (u16)(u >> 16);
}

// One workgroup processes a [TILE_T x 128] output tile for one batch element.
// Wave w (of 8) = (branch = w>>1, co-tile = w&1) for stage 1, feature-tile w for stage 2.
template<int KW, int DIL>
__device__ void tile_worker(const float* __restrict__ x,
                            const float* __restrict__ wbr,
                            const float* __restrict__ bbr,
                            const float* __restrict__ wout,
                            const float* __restrict__ bout,
                            float* __restrict__ out,
                            int b, int t0, int cotile, int br,
                            u16* xbf, u16* xcb)
{
  const int tid  = threadIdx.x;
  const int lane = tid & 31;
  const int ln   = lane & 15;     // N / M index within fragment
  const int hi   = lane >> 4;     // which half-lane group
  const int wid  = tid >> 5;

  // fp32 staging buffer aliases the xc buffer (xc is dead until stage 1)
  float* stg = (float*)xcb;
  const u32 stg_base = (u32)(uintptr_t)stg;  // low 32 bits of flat addr == LDS offset

  // ---- async DMA of x tile (fp32) into LDS staging, 96 rows per phase ----
  auto issue_async = [&](int p) {
    #pragma unroll
    for (int i = 0; i < 6; ++i) {
      const int c   = tid + i * 256;        // 16B chunk id within phase, 0..1535
      const int row = p * 96 + (c >> 4);
      const int t   = t0 - HALO + row;
      if (t >= 0) {                          // causal left pad handled in convert()
        const u32 lds = stg_base + (u32)(c << 4);
        const u64 ga  = (u64)(uintptr_t)(x + (((size_t)b * T_LEN + t) * 64 + ((c & 15) << 2)));
        asm volatile("global_load_async_to_lds_b128 %0, %1, off"
                     :: "v"(lds), "v"(ga) : "memory");
      }
    }
  };
  // ---- fp32 staging -> bf16 padded x tile ----
  auto convert = [&](int p) {
    #pragma unroll
    for (int i = 0; i < 6; ++i) {
      const int c   = tid + i * 256;
      const int row = p * 96 + (c >> 4);
      const int t   = t0 - HALO + row;
      const int ch0 = (c & 15) << 2;
      uint2 pk;
      if (t >= 0) {
        uint4 raw = ((const uint4*)stg)[c];
        pk.x = (u32)f2bf(__uint_as_float(raw.x)) | ((u32)f2bf(__uint_as_float(raw.y)) << 16);
        pk.y = (u32)f2bf(__uint_as_float(raw.z)) | ((u32)f2bf(__uint_as_float(raw.w)) << 16);
      } else {
        pk.x = 0u; pk.y = 0u;                // causal zero padding
      }
      *(uint2*)&xbf[row * XS + ch0] = pk;
    }
  };

  issue_async(0);

  // While DMA flies: per-wave branch-weight B fragments -> registers.
  // B layout (32x16 bf16): lane n holds N = n%16, element e holds K = (e&7) + 8*(n>=16) + 16*(e>=8).
  const int co0 = cotile * 16;
  v16bf bw[KW * 2];
  #pragma unroll
  for (int j = 0; j < KW; ++j) {
    const int r = KW - 1 - j;                // weight tap for time offset j*DIL
    #pragma unroll
    for (int h = 0; h < 2; ++h) {            // Cin half (0..31 / 32..63)
      FragBF fu;
      #pragma unroll
      for (int e = 0; e < 16; ++e)
        fu.u[e] = f2bf(wbr[((r * 64) + h * 32 + hi * 8 + (e & 7) + ((e >> 3) << 4)) * 32 + co0 + ln]);
      bw[j * 2 + h] = fu.v;
    }
  }
  const float bias1 = bbr[co0 + ln];

  asm volatile("s_wait_asynccnt 0" ::: "memory");
  __syncthreads();
  convert(0);
  __syncthreads();                           // staging reads done -> safe to refill
  issue_async(1);
  asm volatile("s_wait_asynccnt 0" ::: "memory");
  __syncthreads();
  convert(1);
  __syncthreads();                           // bf16 x tile complete

  // ---- Stage 1: dilated causal convs as WMMA over (tap, Cin-half) ----
  // A layout (16x32 bf16): lane L row M=L%16; e<8 -> K = e + 8*(L>=16), e>=8 -> K = e+8+8*(L>=16)
  //   => two contiguous 8-bf16 (16B) chunks at ch and ch+16.
  const int cog = br * 32 + co0 + ln;        // concat channel for this lane
  #pragma unroll 1
  for (int mt = 0; mt < 8; ++mt) {
    v8f acc = {0.f,0.f,0.f,0.f,0.f,0.f,0.f,0.f};
    const int baseRow = HALO + mt * 16 + ln;
    #pragma unroll
    for (int j = 0; j < KW; ++j) {
      const int row = baseRow - DIL * j;     // x[t - d*j]
      #pragma unroll
      for (int h = 0; h < 2; ++h) {
        FragBF a;
        const int ch = h * 32 + hi * 8;
        a.q[0] = *(const uint4*)&xbf[row * XS + ch];
        a.q[1] = *(const uint4*)&xbf[row * XS + ch + 16];
        acc = __builtin_amdgcn_wmma_f32_16x16x32_bf16(
                  false, a.v, false, bw[j * 2 + h], (short)0, acc, false, false);
      }
    }
    // C layout: lane n -> N=n%16, VGPR g -> M = g + 8*(n>=16). ReLU+bias -> bf16 xc.
    #pragma unroll
    for (int g = 0; g < 8; ++g) {
      float v = acc[g] + bias1;
      v = v > 0.f ? v : 0.f;
      xcb[(mt * 16 + g + hi * 8) * XCS + cog] = f2bf(v);
    }
  }

  // Stage-2 B fragments (1x1 projection) -> registers, while slower waves finish stage 1.
  const int f0 = wid * 16;
  v16bf bo[4];
  #pragma unroll
  for (int kc = 0; kc < 4; ++kc) {
    FragBF fu;
    #pragma unroll
    for (int e = 0; e < 16; ++e)
      fu.u[e] = f2bf(wout[(kc * 32 + hi * 8 + (e & 7) + ((e >> 3) << 4)) * 128 + f0 + ln]);
    bo[kc] = fu.v;
  }
  const float bias2 = bout[f0 + ln];

  __syncthreads();                           // xc tile complete

  // ---- Stage 2: out = xc @ w_out + b_out (K=128 in 4 WMMA chunks) ----
  #pragma unroll 1
  for (int mt = 0; mt < 8; ++mt) {
    v8f acc = {0.f,0.f,0.f,0.f,0.f,0.f,0.f,0.f};
    const int rowb = (mt * 16 + ln) * XCS;
    #pragma unroll
    for (int kc = 0; kc < 4; ++kc) {
      FragBF a;
      const int ch = kc * 32 + hi * 8;
      a.q[0] = *(const uint4*)&xcb[rowb + ch];
      a.q[1] = *(const uint4*)&xcb[rowb + ch + 16];
      acc = __builtin_amdgcn_wmma_f32_16x16x32_bf16(
                false, a.v, false, bo[kc], (short)0, acc, false, false);
    }
    float* op = out + ((size_t)b * T_LEN + t0 + mt * 16 + hi * 8) * 128 + f0 + ln;
    #pragma unroll
    for (int g = 0; g < 8; ++g)
      op[(size_t)g * 128] = acc[g] + bias2;
  }
}

__global__ __launch_bounds__(256) void msTCN_fused(
    const float* __restrict__ x,
    const float* __restrict__ w_s,  const float* __restrict__ b_s,
    const float* __restrict__ w_m,  const float* __restrict__ b_m,
    const float* __restrict__ w_l,  const float* __restrict__ b_l,
    const float* __restrict__ w_xl, const float* __restrict__ b_xl,
    const float* __restrict__ w_out,const float* __restrict__ b_out,
    float* __restrict__ out)
{
  __shared__ __align__(16) u16 xbf[ROWS * XS];     // 27,648 B  bf16 x tile (+halo)
  __shared__ __align__(16) u16 xcb[TILE_T * XCS];  // 34,816 B  bf16 xc tile / fp32 staging

  const int blk = blockIdx.x;
  const int b   = blk >> 6;                 // 64 tiles per batch (8192/128)
  const int t0  = (blk & 63) * TILE_T;
  const int wid = threadIdx.x >> 5;
  const int br  = wid >> 1;
  const int ct  = wid & 1;

  switch (br) {
    case 0:  tile_worker<3, 1>(x, w_s,  b_s,  w_out, b_out, out, b, t0, ct, 0, xbf, xcb); break;
    case 1:  tile_worker<5, 2>(x, w_m,  b_m,  w_out, b_out, out, b, t0, ct, 1, xbf, xcb); break;
    case 2:  tile_worker<9, 4>(x, w_l,  b_l,  w_out, b_out, out, b, t0, ct, 2, xbf, xcb); break;
    default: tile_worker<9, 8>(x, w_xl, b_xl, w_out, b_out, out, b, t0, ct, 3, xbf, xcb); break;
  }
}

extern "C" void kernel_launch(void* const* d_in, const int* in_sizes, int n_in,
                              void* d_out, int out_size, void* d_ws, size_t ws_size,
                              hipStream_t stream) {
  (void)in_sizes; (void)n_in; (void)d_ws; (void)ws_size; (void)out_size;
  const float* x    = (const float*)d_in[0];
  const float* w_s  = (const float*)d_in[1];
  const float* b_s  = (const float*)d_in[2];
  const float* w_m  = (const float*)d_in[3];
  const float* b_m  = (const float*)d_in[4];
  const float* w_l  = (const float*)d_in[5];
  const float* b_l  = (const float*)d_in[6];
  const float* w_xl = (const float*)d_in[7];
  const float* b_xl = (const float*)d_in[8];
  const float* w_o  = (const float*)d_in[9];
  const float* b_o  = (const float*)d_in[10];
  float* out = (float*)d_out;

  dim3 grid(16 * (T_LEN / TILE_T));   // 1024 workgroups
  dim3 block(256);                    // 8 wave32 waves
  msTCN_fused<<<grid, block, 0, stream>>>(x, w_s, b_s, w_m, b_m, w_l, b_l,
                                          w_xl, b_xl, w_o, b_o, out);
}
